// WindowAttention_49143015801517
// MI455X (gfx1250) — matrix operand order repaired and checked by reading, not compiled, not run on hardware
//
#include <hip/hip_runtime.h>
#include <math.h>
#include <stdint.h>

typedef _Float16 half_t;
typedef __attribute__((ext_vector_type(16))) _Float16 v16h;
typedef __attribute__((ext_vector_type(8)))  float    v8f;
typedef __attribute__((ext_vector_type(4)))  float    v4f;

#define WS7   7
#define NTOK  49           // tokens per window
#define NPAD  64           // padded to 4 M-tiles of 16
#define CDIM  96
#define NH    3
#define HD    32
#define QK_N  288          // 3*CDIM
#define SIMLD 50           // sim leading dim (only cols < 49 stored)
#define SCALE 0.17677669529663687f   // 32^-0.5

// ---------------- fragment helpers (CDNA5 wave32 WMMA f16 layouts) ----------

// A matrix 16x32 f16: lane holds row = row0+(lane&15); K interleave:
// VGPR v in 0..3 -> K = (lane>>4)*8 + 2v ; v in 4..7 -> K = 16 + (lane>>4)*8 + 2(v-4)
__device__ __forceinline__ v16h load_frag_a(const half_t* __restrict__ base,
                                            int ld, int row0, int k0, int lane) {
  const int row = row0 + (lane & 15);
  const int kg  = (lane >> 4) << 3;
  const half_t* p = base + row * ld + k0;
  v16h a;
#pragma unroll
  for (int v = 0; v < 8; ++v) {
    const int kb = ((v & 4) ? 16 : 0) + kg + ((v & 3) << 1);
    a[2 * v]     = p[kb];
    a[2 * v + 1] = p[kb + 1];
  }
  return a;
}

// Same A fragment, sourced from f32 LDS staging (64 rows, padded rows are
// zeroed) with on-the-fly cvt. Branchless: batch all loads, then convert.
__device__ __forceinline__ v16h load_frag_a_f32(const float* __restrict__ base,
                                                int ld, int row0, int k0, int lane) {
  const int row = row0 + (lane & 15);
  const int kg  = (lane >> 4) << 3;
  const float* p = base + row * ld + k0;
  float f[16];
#pragma unroll
  for (int v = 0; v < 8; ++v) {
    const int kb = ((v & 4) ? 16 : 0) + kg + ((v & 3) << 1);
    f[2 * v]     = p[kb];
    f[2 * v + 1] = p[kb + 1];
  }
  v16h a;
#pragma unroll
  for (int i = 0; i < 16; ++i) a[i] = (half_t)f[i];
  return a;
}

// B matrix 32x16 f16: lane holds col = col0+(lane&15); same K interleave as A:
// elements 0..7 -> K = (lane>>4)*8 + s ; elements 8..15 -> K = 16 + (lane>>4)*8 + (s-8).
__device__ __forceinline__ v16h load_frag_b(const half_t* __restrict__ base,
                                            int ld, int col0, int k0, int lane) {
  const int col = col0 + (lane & 15);
  const int kh  = (lane >> 4) << 3;
  const half_t* p = base + col * ld + k0 + kh;
  v16h b;
#pragma unroll
  for (int v = 0; v < 8; ++v) {
    const int kb = ((v & 4) ? 16 : 0) + ((v & 3) << 1);
    b[2 * v]     = p[kb];
    b[2 * v + 1] = p[kb + 1];
  }
  return b;
}

#define WMMA_F16(acc, a, b) \
  __builtin_amdgcn_wmma_f32_16x16x32_f16(false, (a), false, (b), (short)0, (acc), false, false)

// ---------------- prep kernel: weight transpose->f16, bias table ------------
__global__ void prep_kernel(const float* __restrict__ w_qkv,
                            const float* __restrict__ w_out,
                            const float* __restrict__ rel_pos,
                            half_t* __restrict__ wq_t,   // [288][96] f16 (N-major, K contiguous)
                            half_t* __restrict__ wo_t,   // [96][96]  f16
                            float*  __restrict__ bias)   // [3][49][49]
{
  const int NWQ = QK_N * CDIM;       // 27648
  const int NWO = CDIM * CDIM;       // 9216
  const int NB  = NH * NTOK * NTOK;  // 7203
  int i = blockIdx.x * blockDim.x + threadIdx.x;
  if (i < NWQ) {
    int n = i / CDIM, k = i - n * CDIM;
    wq_t[n * CDIM + k] = (half_t)w_qkv[k * QK_N + n];
  } else if (i < NWQ + NWO) {
    int j = i - NWQ;
    int n = j / CDIM, k = j - n * CDIM;
    wo_t[n * CDIM + k] = (half_t)w_out[k * CDIM + n];
  } else if (i < NWQ + NWO + NB) {
    int j = i - NWQ - NWO;
    int g = j / (NTOK * NTOK);
    int rem = j - g * (NTOK * NTOK);
    int a = rem / NTOK, b = rem - a * NTOK;
    int yi = a / WS7, xi = a - yi * WS7;
    int yj = b / WS7, xj = b - yj * WS7;
    int ridx = (yi - yj + WS7 - 1) * (2 * WS7 - 1) + (xi - xj + WS7 - 1);
    bias[j] = rel_pos[g * (2 * WS7 - 1) * (2 * WS7 - 1) + ridx];
  }
}

// ---------------- fused window-attention kernel -----------------------------
__launch_bounds__(256)
__global__ void win_attn_kernel(const float* __restrict__ x,
                                const float* __restrict__ b_qkv,
                                const float* __restrict__ b_out,
                                const half_t* __restrict__ wq_t,
                                const half_t* __restrict__ wo_t,
                                const float* __restrict__ bias,
                                float* __restrict__ out)
{
  // pool: f32 window staging (64x96 = 24576 B, rows >= 49 zeroed) in stage 0/1,
  //       reused as f16 att-out (64x96 = 12288 B) in stage 4/5.
  __shared__ __align__(16) char pool[NPAD * CDIM * 4];
  __shared__ half_t qs[NH][NPAD][HD];           // q * SCALE
  __shared__ half_t ks[NH][NPAD][HD];           // k (token-major, d contiguous -> B frags)
  __shared__ half_t vt[NH][HD][NPAD];           // v transposed (d-major, token contiguous)
  __shared__ float  sim[NH][NPAD][SIMLD];       // logits (cols < 49 only)
  __shared__ half_t att[NH][NPAD][NPAD];        // softmax probs (padded cols/rows = 0)

  float*  xraw = (float*)pool;                  // [64][96] f32
  half_t* ao   = (half_t*)pool;                 // [64][96] f16

  const int tid  = threadIdx.x;
  const int lane = tid & 31;
  const int wave = tid >> 5;                    // 0..7 (wave32)

  const int blk  = blockIdx.x;                  // b*64 + hwin*8 + wwin
  const int wcol = blk & 7;
  const int wrow = (blk >> 3) & 7;
  const int bimg = blk >> 6;
  const float* xb = x + (size_t)bimg * 56 * 56 * CDIM;

  // ---- Stage 0a: zero the 15 padding rows (49..63) with b128 stores ----
  {
    v4f z = {};
    v4f* xq = (v4f*)(xraw + NTOK * CDIM);       // 16-B aligned (18816 % 16 == 0)
    for (int i = tid; i < (NPAD - NTOK) * (CDIM / 4); i += 256) xq[i] = z;
  }

  // ---- Stage 0b: async gather of the strided window into LDS staging ----
  // 49 rows x 384 B, each row = 24 aligned 16-B chunks -> 1176 async B128 ops.
  for (int c = tid; c < NTOK * (CDIM / 4); c += 256) {
    const int r  = c / (CDIM / 4);
    const int cc = (c - r * (CDIM / 4)) * 4;            // float column
    const int p  = r / WS7, q = r - p * WS7;
    const float* gp = xb + ((size_t)(p * 8 + wrow) * 56 + (q * 8 + wcol)) * CDIM + cc;
    const unsigned lds_off = (unsigned)(uintptr_t)(xraw + r * CDIM + cc);
    asm volatile("global_load_async_to_lds_b128 %0, %1, off"
                 :: "v"(lds_off), "v"((unsigned long long)(uintptr_t)gp)
                 : "memory");
  }
#if __has_builtin(__builtin_amdgcn_s_wait_asynccnt)
  __builtin_amdgcn_s_wait_asynccnt(0);
#else
  asm volatile("s_wait_asynccnt 0x0" ::: "memory");
#endif
  __syncthreads();

  // ---- Stage 1: QKV GEMM (64x96 @ 96x288), 72 tiles / 8 waves ----
  // A fragments come straight from the f32 staging buffer (cvt on load);
  // m is wave-invariant, yet the 3 A fragments are rebuilt for every tile.
  for (int t = wave; t < 72; t += 8) {
    const int m = (t & 3) << 4;
    const int n = (t >> 2) << 4;
    v8f acc = {};
#pragma unroll
    for (int kk = 0; kk < 3; ++kk) {
      v16h a = load_frag_a_f32(xraw, CDIM, m, kk * 32, lane);
      v16h b = load_frag_b(wq_t, CDIM, n, kk * 32, lane);
      acc = WMMA_F16(acc, a, b);
    }
    // section / head / base-offset are tile-uniform -> scalar branches
    const int sec = n / CDIM;           // 0=q 1=k 2=v
    const int hb  = n - sec * CDIM;     // 0..80, 16-aligned
    const int hd  = hb >> 5;
    const int dd  = (hb & 31) + (lane & 15);
    const float bq = b_qkv[n + (lane & 15)];
    if (sec == 0) {
#pragma unroll
      for (int r = 0; r < 8; ++r) {
        const int row = m + r + ((lane >> 4) << 3);
        qs[hd][row][dd] = (half_t)((acc[r] + bq) * SCALE);
      }
    } else if (sec == 1) {
#pragma unroll
      for (int r = 0; r < 8; ++r) {
        const int row = m + r + ((lane >> 4) << 3);
        ks[hd][row][dd] = (half_t)(acc[r] + bq);
      }
    } else {
#pragma unroll
      for (int r = 0; r < 8; ++r) {
        const int row = m + r + ((lane >> 4) << 3);
        vt[hd][dd][row] = (half_t)(acc[r] + bq);
      }
    }
  }
  __syncthreads();

  // ---- Stage 2: sim = (q*scale) @ k^T + bias ; 48 tiles / 8 waves ----
  for (int t = wave; t < 48; t += 8) {
    const int hd = t >> 4;
    const int r2 = t & 15;
    const int m = (r2 >> 2) << 4;
    const int n = (r2 & 3) << 4;
    v8f acc = {};
    v16h a = load_frag_a(&qs[hd][0][0], HD, m, 0, lane);
    v16h b = load_frag_b(&ks[hd][0][0], HD, n, 0, lane);
    acc = WMMA_F16(acc, a, b);
    const int col = n + (lane & 15);
    if (col < NTOK) {
      const float* bp = bias + hd * NTOK * NTOK + col;
#pragma unroll
      for (int r = 0; r < 8; ++r) {
        const int row = m + r + ((lane >> 4) << 3);
        float v = acc[r];
        if (row < NTOK) v += bp[row * NTOK];
        sim[hd][row][col] = v;
      }
    }
  }
  __syncthreads();

  // ---- Stage 3: row softmax (147 active threads) + zero padded att rows ----
  if (tid < NH * NTOK) {
    const int hd = tid / NTOK;
    const int r  = tid - hd * NTOK;
    float mx = -1e30f;
    for (int c = 0; c < NTOK; ++c) mx = fmaxf(mx, sim[hd][r][c]);
    float s = 0.0f;
    for (int c = 0; c < NTOK; ++c) s += __expf(sim[hd][r][c] - mx);
    const float inv = 1.0f / s;
    for (int c = 0; c < NTOK; ++c)
      att[hd][r][c] = (half_t)(__expf(sim[hd][r][c] - mx) * inv);
    for (int c = NTOK; c < NPAD; ++c) att[hd][r][c] = (half_t)0.0f;
  }
  for (int i = tid; i < NH * (NPAD - NTOK) * NPAD; i += 256) {
    int hd = i / ((NPAD - NTOK) * NPAD);
    int rem = i - hd * (NPAD - NTOK) * NPAD;
    int r = NTOK + rem / NPAD;
    int c = rem - (rem / NPAD) * NPAD;
    att[hd][r][c] = (half_t)0.0f;
  }
  __syncthreads();

  // ---- Stage 4: att @ V -> ao (64x96 f16, aliases staging pool) ----
  for (int t = wave; t < 24; t += 8) {
    const int hd = t >> 3;
    const int r2 = t & 7;
    const int m = (r2 >> 1) << 4;
    const int n = (r2 & 1) << 4;
    v8f acc = {};
#pragma unroll
    for (int kk = 0; kk < 2; ++kk) {
      v16h a = load_frag_a(&att[hd][0][0], NPAD, m, kk * 32, lane);
      v16h b = load_frag_b(&vt[hd][0][0], NPAD, n, kk * 32, lane);
      acc = WMMA_F16(acc, a, b);
    }
    const int col = hd * HD + n + (lane & 15);
#pragma unroll
    for (int r = 0; r < 8; ++r) {
      const int row = m + r + ((lane >> 4) << 3);
      ao[row * CDIM + col] = (half_t)acc[r];
    }
  }
  __syncthreads();

  // ---- Stage 5: out = att_out @ w_out + b_out, scatter to window ----
  float* ob = out + (size_t)bimg * 56 * 56 * CDIM;
  for (int t = wave; t < 24; t += 8) {
    const int m = (t / 6) << 4;
    const int n = (t % 6) << 4;
    v8f acc = {};
#pragma unroll
    for (int kk = 0; kk < 3; ++kk) {
      v16h a = load_frag_a(ao, CDIM, m, kk * 32, lane);
      v16h b = load_frag_b(wo_t, CDIM, n, kk * 32, lane);
      acc = WMMA_F16(acc, a, b);
    }
    const int coln = n + (lane & 15);
    const float bo = b_out[coln];
#pragma unroll
    for (int r = 0; r < 8; ++r) {
      const int row = m + r + ((lane >> 4) << 3);
      if (row < NTOK) {
        int p = row / WS7, q = row - p * WS7;
        ob[((size_t)(p * 8 + wrow) * 56 + (q * 8 + wcol)) * CDIM + coln] = acc[r] + bo;
      }
    }
  }
}

// ---------------- host-side launcher ---------------------------------------
extern "C" void kernel_launch(void* const* d_in, const int* in_sizes, int n_in,
                              void* d_out, int out_size, void* d_ws, size_t ws_size,
                              hipStream_t stream) {
  const float* x       = (const float*)d_in[0];
  const float* w_qkv   = (const float*)d_in[1];
  const float* b_qkv   = (const float*)d_in[2];
  const float* rel_pos = (const float*)d_in[3];
  const float* w_out   = (const float*)d_in[4];
  const float* b_out   = (const float*)d_in[5];
  float* out = (float*)d_out;

  char* ws = (char*)d_ws;
  half_t* wq_t = (half_t*)(ws);                    // 288*96*2  = 55296 B
  half_t* wo_t = (half_t*)(ws + 55296);            // 96*96*2   = 18432 B
  float*  bias = (float*)(ws + 55296 + 18432);     // 3*49*49*4 = 28812 B

  const int prep_n = QK_N * CDIM + CDIM * CDIM + NH * NTOK * NTOK; // 44067
  prep_kernel<<<(prep_n + 255) / 256, 256, 0, stream>>>(
      w_qkv, w_out, rel_pos, wq_t, wo_t, bias);

  win_attn_kernel<<<dim3(64 * 8 * 8), dim3(256), 0, stream>>>(
      x, b_qkv, b_out, wq_t, wo_t, bias, out);
}
